// BHS_GCN_DQN_16724602651185
// MI455X (gfx1250) — matrix operand order repaired and checked by reading, compiled, not run on hardware
//
#include <hip/hip_runtime.h>
#include <hip/hip_bf16.h>
#include <stdint.h>

// ---------------------------------------------------------------------------
// Problem constants (from the reference)
// ---------------------------------------------------------------------------
constexpr int B_SZ   = 32;
constexpr int N_NODE = 4096;
constexpr int NT     = B_SZ * N_NODE;   // 131072 nodes
constexpr int EDG    = 1048576;         // edges
constexpr int FEATK  = N_NODE * 128;    // 524288 (head GEMM K)

// ---------------------------------------------------------------------------
// Types for WMMA (gfx1250, wave32)
// ---------------------------------------------------------------------------
typedef __attribute__((ext_vector_type(16))) __bf16 v16bf;
typedef __attribute__((ext_vector_type(8)))  __bf16 bf16x8;
typedef __attribute__((ext_vector_type(8)))  float  v8f;

union BF16Frag { v16bf v; bf16x8 h[2]; };

__device__ inline void atomAddF(float* p, float v) {
  // relaxed, device scope -> native global_atomic_add_f32 on gfx1250
  __hip_atomic_fetch_add(p, v, __ATOMIC_RELAXED, __HIP_MEMORY_SCOPE_AGENT);
}

// ---------------------------------------------------------------------------
// gcn_norm pipeline
// ---------------------------------------------------------------------------
__global__ void k_init_deg(float* deg) {
  int i = blockIdx.x * 256 + threadIdx.x;
  if (i < NT) deg[i] = 1.0f;   // self-loop weight
}

__global__ void k_deg_accum(const int* __restrict__ col,
                            const float* __restrict__ ew,
                            float* __restrict__ deg) {
  int e = blockIdx.x * 256 + threadIdx.x;
  if (e < EDG) atomAddF(&deg[col[e]], ew[e]);
}

__global__ void k_dinv(const float* __restrict__ deg, float* __restrict__ dinv) {
  int i = blockIdx.x * 256 + threadIdx.x;
  if (i < NT) {
    float d = deg[i];
    dinv[i] = (d > 0.f) ? rsqrtf(fmaxf(d, 1e-30f)) : 0.f;
  }
}

__global__ void k_edge_norm(const int* __restrict__ row, const int* __restrict__ col,
                            const float* __restrict__ ew, const float* __restrict__ dinv,
                            float* __restrict__ nrm) {
  int e = blockIdx.x * 256 + threadIdx.x;
  if (e < EDG) nrm[e] = dinv[row[e]] * ew[e] * dinv[col[e]];
}

// ---------------------------------------------------------------------------
// Weight packing: W[K][N] f32 -> Wt[N][Kp] bf16 (transposed + zero K-padding)
// ---------------------------------------------------------------------------
__global__ void k_pack_w(const float* __restrict__ W, __bf16* __restrict__ Wt,
                         int K, int Kp, int Nout) {
  int idx = blockIdx.x * 256 + threadIdx.x;
  if (idx >= Nout * Kp) return;
  int n = idx / Kp, k = idx % Kp;
  Wt[idx] = (k < K) ? (__bf16)W[k * Nout + n] : (__bf16)0.f;
}

// Head weights: W[FEATK][N] f32 -> Wt[N][FEATK] bf16 (K = 2^19, hardcoded shift)
__global__ void k_pack_head(const float* __restrict__ W, __bf16* __restrict__ Wt,
                            int Nout) {
  size_t idx = (size_t)blockIdx.x * 256 + threadIdx.x;
  if (idx >= (size_t)Nout * FEATK) return;
  int    n = (int)(idx >> 19);
  size_t k = idx & (FEATK - 1);
  Wt[idx] = (__bf16)W[k * Nout + n];
}

__global__ void k_copy_f32(const float* __restrict__ s, float* __restrict__ d, int n) {
  int i = blockIdx.x * 256 + threadIdx.x;
  if (i < n) d[i] = s[i];
}

// ---------------------------------------------------------------------------
// Node GEMM: xw[M,Nout] = (relu?)A[M,K] @ W[K,Nout], bf16 WMMA, f32 accum.
// All shapes are template params -> accumulators stay in VGPRs (no scratch).
// Block = 256 threads = 8 waves; block computes 128 rows x Nout.
// ---------------------------------------------------------------------------
template <int K, int Kp, int Nout, int RELU>
__global__ __launch_bounds__(256)
void k_gemm_wmma(const float* __restrict__ A, const __bf16* __restrict__ Wt,
                 float* __restrict__ xw) {
  constexpr int STRIDE = Kp + 8;          // multiple of 8 -> 16B-aligned frags
  constexpr int NTILES = Nout / 16;
  __shared__ __bf16 As[128 * STRIDE];

  const int blockRow = blockIdx.x * 128;
  const int tid      = threadIdx.x;

  // Stage + convert (+ optional fused ReLU) + zero-pad K..Kp
  #pragma unroll 4
  for (int idx = tid; idx < 128 * Kp; idx += 256) {
    int r = idx / Kp, c = idx % Kp;
    float v = 0.f;
    if (c < K) {
      v = A[(size_t)(blockRow + r) * K + c];
      if (RELU) v = fmaxf(v, 0.f);
    }
    As[r * STRIDE + c] = (__bf16)v;
  }
  __syncthreads();

  const int wave = tid >> 5, lane = tid & 31;
  const int half = lane >> 4, l16 = lane & 15;
  const int rbase = wave * 16;

  v8f acc[NTILES];
  #pragma unroll
  for (int t = 0; t < NTILES; ++t)
    #pragma unroll
    for (int i = 0; i < 8; ++i) acc[t][i] = 0.f;

  #pragma unroll
  for (int k0 = 0; k0 < Kp; k0 += 32) {
    // A frag (16x32 bf16): half 0 -> K[0..8)+[16..24), half 1 -> K[8..16)+[24..32)
    BF16Frag a;
    const __bf16* ap = &As[(rbase + l16) * STRIDE + k0 + half * 8];
    a.h[0] = *(const bf16x8*)ap;
    a.h[1] = *(const bf16x8*)(ap + 16);

    #pragma unroll
    for (int t = 0; t < NTILES; ++t) {
      // B frag (32x16 bf16): lane = col l16, half 0 -> K[0..8)+[16..24), half 1 -> K[8..16)+[24..32)
      BF16Frag b;
      const __bf16* bp = Wt + (size_t)(t * 16 + l16) * Kp + k0 + half * 8;
      b.h[0] = *(const bf16x8*)bp;
      b.h[1] = *(const bf16x8*)(bp + 16);
      acc[t] = __builtin_amdgcn_wmma_f32_16x16x32_bf16(
          false, a.v, false, b.v, (short)0, acc[t], false, false);
    }
  }

  // C/D layout: VGPR i -> row (half*8 + i), col l16
  #pragma unroll
  for (int t = 0; t < NTILES; ++t) {
    int col = t * 16 + l16;
    size_t base = (size_t)(blockRow + rbase + half * 8) * Nout + col;
    #pragma unroll
    for (int i = 0; i < 8; ++i)
      xw[base + (size_t)i * Nout] = acc[t][i];
  }
}

// ---------------------------------------------------------------------------
// Aggregation: out = bias + dinv^2 * xw (self loop), then edge scatter-add
// ---------------------------------------------------------------------------
__global__ void k_seed(const float* __restrict__ xw, const float* __restrict__ dinv,
                       const float* __restrict__ bias, float* __restrict__ out) {
  int i = blockIdx.x, j = threadIdx.x;                 // grid = NT, block = W
  size_t idx = (size_t)i * blockDim.x + j;
  float di = dinv[i];
  out[idx] = bias[j] + di * di * xw[idx];
}

template <int W>
__global__ __launch_bounds__(256)
void k_scatter(const float* __restrict__ xw, float* __restrict__ out,
               const int* __restrict__ row, const int* __restrict__ col,
               const float* __restrict__ nrm) {
  constexpr unsigned CPE = W / 4;                      // float4 chunks per edge
  unsigned idx = blockIdx.x * 256 + threadIdx.x;
  unsigned e = idx / CPE;
  if (e >= EDG) return;
  int c = (int)(idx % CPE) * 4;
  int r = row[e], cl = col[e];
  float nm = nrm[e];
  const float4 v = *(const float4*)(xw + (size_t)r * W + c);
  float* o = out + (size_t)cl * W + c;
  atomAddF(o + 0, nm * v.x);
  atomAddF(o + 1, nm * v.y);
  atomAddF(o + 2, nm * v.z);
  atomAddF(o + 3, nm * v.w);
}

// Final activation after conv5 -> bf16 feature matrix [32][FEATK]
__global__ void k_feat_cvt(const float* __restrict__ act, __bf16* __restrict__ featb) {
  size_t idx = (size_t)blockIdx.x * blockDim.x + threadIdx.x;  // NT x 128
  featb[idx] = (__bf16)fmaxf(act[idx], 0.f);
}

// ---------------------------------------------------------------------------
// Head GEMM: out[32,Nout] += feat[32,K=FEATK] @ WtT, split-K WMMA + atomics.
// Grid = K-slabs; each wave covers a contiguous K range, accumulates
// 2 x (Nout/16) 16x16 tiles in VGPRs, then atomically reduces into out.
// ---------------------------------------------------------------------------
template <int Nout, int KPB>
__global__ __launch_bounds__(256)
void k_head_gemm(const __bf16* __restrict__ feat, const __bf16* __restrict__ Wt,
                 float* __restrict__ out) {
  constexpr int K      = FEATK;
  constexpr int NTILES = Nout / 16;       // 2 or 4
  constexpr int KPW    = KPB / 8;         // per-wave contiguous K range

  const int tid = threadIdx.x;
  const int wave = tid >> 5, lane = tid & 31;
  const int half = lane >> 4, l16 = lane & 15;
  const int kStart = blockIdx.x * KPB + wave * KPW;

  v8f acc[2][NTILES];
  #pragma unroll
  for (int m = 0; m < 2; ++m)
    #pragma unroll
    for (int t = 0; t < NTILES; ++t)
      #pragma unroll
      for (int i = 0; i < 8; ++i) acc[m][t][i] = 0.f;

  for (int k0 = kStart; k0 < kStart + KPW; k0 += 32) {
    BF16Frag a[2];
    #pragma unroll
    for (int m = 0; m < 2; ++m) {
      const __bf16* ap = feat + (size_t)(m * 16 + l16) * K + k0 + half * 8;
      a[m].h[0] = *(const bf16x8*)ap;
      a[m].h[1] = *(const bf16x8*)(ap + 16);
    }
    #pragma unroll
    for (int t = 0; t < NTILES; ++t) {
      const __bf16* bp = Wt + (size_t)(t * 16 + l16) * K + k0 + half * 8;
      __builtin_prefetch((const void*)(bp + 256), 0, 0);   // global_prefetch_b8
      BF16Frag b;
      b.h[0] = *(const bf16x8*)bp;
      b.h[1] = *(const bf16x8*)(bp + 16);
      acc[0][t] = __builtin_amdgcn_wmma_f32_16x16x32_bf16(
          false, a[0].v, false, b.v, (short)0, acc[0][t], false, false);
      acc[1][t] = __builtin_amdgcn_wmma_f32_16x16x32_bf16(
          false, a[1].v, false, b.v, (short)0, acc[1][t], false, false);
    }
  }

  #pragma unroll
  for (int m = 0; m < 2; ++m)
    #pragma unroll
    for (int t = 0; t < NTILES; ++t) {
      int col = t * 16 + l16;
      #pragma unroll
      for (int i = 0; i < 8; ++i) {
        int rowi = m * 16 + half * 8 + i;
        atomAddF(&out[(size_t)rowi * Nout + col], acc[m][t][i]);
      }
    }
}

// ---------------------------------------------------------------------------
// Small head kernels
// ---------------------------------------------------------------------------
__global__ void k_seed_small(float* __restrict__ out, const float* __restrict__ bias,
                             int rows, int N) {
  int i = blockIdx.x * 256 + threadIdx.x;
  if (i < rows * N) out[i] = bias[i % N];
}

__global__ void k_relu_inplace(float* __restrict__ x, int n) {
  int i = blockIdx.x * 256 + threadIdx.x;
  if (i < n) x[i] = fmaxf(x[i], 0.f);
}

__global__ void k_v2_dense(const float* __restrict__ v1, const float* __restrict__ W,
                           const float* __restrict__ b, float* __restrict__ out) {
  int t = blockIdx.x * 256 + threadIdx.x;      // 32*64
  if (t >= 32 * 64) return;
  int bb = t >> 6, n = t & 63;
  float s = b[n];
  #pragma unroll 8
  for (int k = 0; k < 64; ++k) s += v1[bb * 64 + k] * W[k * 64 + n];
  out[t] = fmaxf(s, 0.f);
}

__global__ void k_v3_dense(const float* __restrict__ v2, const float* __restrict__ W,
                           const float* __restrict__ b, float* __restrict__ val) {
  int t = threadIdx.x;
  if (t >= 32) return;
  float s = b[0];
  #pragma unroll 8
  for (int k = 0; k < 64; ++k) s += v2[t * 64 + k] * W[k];
  val[t] = s;
}

__global__ void k_combine(const float* __restrict__ val, const float* __restrict__ adv,
                          float* __restrict__ out) {
  int t = blockIdx.x * 256 + threadIdx.x;      // 32*8*4
  if (t >= 1024) return;
  int bb = t >> 5, g = (t >> 2) & 7, p = t & 3;
  const float* a = adv + bb * 32 + g * 4;
  float mean = 0.25f * (a[0] + a[1] + a[2] + a[3]);
  out[t] = val[bb] + a[p] - mean;
}

// ---------------------------------------------------------------------------
// Host launcher
// ---------------------------------------------------------------------------
static inline int ceil_div(long long a, long long b) { return (int)((a + b - 1) / b); }

extern "C" void kernel_launch(void* const* d_in, const int* in_sizes, int n_in,
                              void* d_out, int out_size, void* d_ws, size_t ws_size,
                              hipStream_t stream) {
  (void)in_sizes; (void)n_in; (void)out_size; (void)ws_size;

  const float* x     = (const float*)d_in[0];
  const int*   eidx  = (const int*)d_in[1];
  const int*   erow  = eidx;            // edge_index[0]
  const int*   ecol  = eidx + EDG;      // edge_index[1]
  const float* ew    = (const float*)d_in[2];
  const float* W1    = (const float*)d_in[3];   const float* b1  = (const float*)d_in[4];
  const float* W21   = (const float*)d_in[5];   const float* b21 = (const float*)d_in[6];
  const float* W22   = (const float*)d_in[7];   const float* b22 = (const float*)d_in[8];
  const float* W23   = (const float*)d_in[9];   const float* b23 = (const float*)d_in[10];
  const float* W3    = (const float*)d_in[11];  const float* b3  = (const float*)d_in[12];
  const float* W4    = (const float*)d_in[13];  const float* b4  = (const float*)d_in[14];
  const float* W5    = (const float*)d_in[15];  const float* b5  = (const float*)d_in[16];
  const float* advW  = (const float*)d_in[17];  const float* advb = (const float*)d_in[18];
  const float* v1W   = (const float*)d_in[19];  const float* v1b  = (const float*)d_in[20];
  const float* v2W   = (const float*)d_in[21];  const float* v2b  = (const float*)d_in[22];
  const float* v3W   = (const float*)d_in[23];  const float* v3b  = (const float*)d_in[24];
  float* outp = (float*)d_out;

  // ---- carve workspace -------------------------------------------------
  uintptr_t p = (uintptr_t)d_ws;
  auto alloc = [&](size_t bytes) -> void* {
    void* r = (void*)p;
    p += (bytes + 255) & ~(size_t)255;
    return r;
  };
  float*  act   = (float*)alloc((size_t)NT * 192 * 4);      // activations (<=192 wide)
  float*  xw    = (float*)alloc((size_t)NT * 192 * 4);      // pre-aggregation h@W
  __bf16* featb = (__bf16*)alloc((size_t)NT * 128 * 2);     // bf16 features [32][FEATK]
  __bf16* wtadv = (__bf16*)alloc((size_t)32 * FEATK * 2);
  __bf16* wtv1  = (__bf16*)alloc((size_t)64 * FEATK * 2);
  float*  nrm   = (float*)alloc((size_t)EDG * 4);
  float*  deg   = (float*)alloc((size_t)NT * 4);
  float*  dinv  = (float*)alloc((size_t)NT * 4);
  __bf16* wt1   = (__bf16*)alloc(128 * 32 * 2);             // [128][Kp=32]
  __bf16* wt2   = (__bf16*)alloc(192 * 128 * 2);            // merged W21|W22|W23
  __bf16* wt3   = (__bf16*)alloc(128 * 192 * 2);
  __bf16* wt4   = (__bf16*)alloc(128 * 128 * 2);
  __bf16* wt5   = (__bf16*)alloc(128 * 128 * 2);
  float*  b2cat = (float*)alloc(192 * 4);
  float*  advp  = (float*)alloc(32 * 32 * 4);
  float*  v1a   = (float*)alloc(32 * 64 * 4);
  float*  v2a   = (float*)alloc(32 * 64 * 4);
  float*  valb  = (float*)alloc(32 * 4);

  // ---- pack weights (bf16, transposed, K-padded) -----------------------
  k_pack_w<<<ceil_div(128 * 32, 256), 256, 0, stream>>>(W1, wt1, 8, 32, 128);
  k_pack_w<<<ceil_div(64 * 128, 256), 256, 0, stream>>>(W21, wt2 + 0 * 64 * 128, 128, 128, 64);
  k_pack_w<<<ceil_div(64 * 128, 256), 256, 0, stream>>>(W22, wt2 + 1 * 64 * 128, 128, 128, 64);
  k_pack_w<<<ceil_div(64 * 128, 256), 256, 0, stream>>>(W23, wt2 + 2 * 64 * 128, 128, 128, 64);
  k_pack_w<<<ceil_div(128 * 192, 256), 256, 0, stream>>>(W3, wt3, 192, 192, 128);
  k_pack_w<<<ceil_div(128 * 128, 256), 256, 0, stream>>>(W4, wt4, 128, 128, 128);
  k_pack_w<<<ceil_div(128 * 128, 256), 256, 0, stream>>>(W5, wt5, 128, 128, 128);
  k_copy_f32<<<1, 256, 0, stream>>>(b21, b2cat + 0, 64);
  k_copy_f32<<<1, 256, 0, stream>>>(b22, b2cat + 64, 64);
  k_copy_f32<<<1, 256, 0, stream>>>(b23, b2cat + 128, 64);
  k_pack_head<<<ceil_div((long long)32 * FEATK, 256), 256, 0, stream>>>(advW, wtadv, 32);
  k_pack_head<<<ceil_div((long long)64 * FEATK, 256), 256, 0, stream>>>(v1W, wtv1, 64);

  // ---- gcn_norm --------------------------------------------------------
  k_init_deg<<<ceil_div(NT, 256), 256, 0, stream>>>(deg);
  k_deg_accum<<<ceil_div(EDG, 256), 256, 0, stream>>>(ecol, ew, deg);
  k_dinv<<<ceil_div(NT, 256), 256, 0, stream>>>(deg, dinv);
  k_edge_norm<<<ceil_div(EDG, 256), 256, 0, stream>>>(erow, ecol, ew, dinv, nrm);

  const int gemmGrid = NT / 128;   // 1024
  const int scat128  = ceil_div((long long)EDG * 32, 256);
  const int scat192  = ceil_div((long long)EDG * 48, 256);

  // ---- layer 1: 8 -> 128 ----------------------------------------------
  k_gemm_wmma<8, 32, 128, 0><<<gemmGrid, 256, 0, stream>>>(x, wt1, xw);
  k_seed<<<NT, 128, 0, stream>>>(xw, dinv, b1, act);
  k_scatter<128><<<scat128, 256, 0, stream>>>(xw, act, erow, ecol, nrm);

  // ---- layer 2 (merged 3x): 128 -> 192 ---------------------------------
  k_gemm_wmma<128, 128, 192, 1><<<gemmGrid, 256, 0, stream>>>(act, wt2, xw);
  k_seed<<<NT, 192, 0, stream>>>(xw, dinv, b2cat, act);
  k_scatter<192><<<scat192, 256, 0, stream>>>(xw, act, erow, ecol, nrm);

  // ---- layer 3: 192 -> 128 ---------------------------------------------
  k_gemm_wmma<192, 192, 128, 1><<<gemmGrid, 256, 0, stream>>>(act, wt3, xw);
  k_seed<<<NT, 128, 0, stream>>>(xw, dinv, b3, act);
  k_scatter<128><<<scat128, 256, 0, stream>>>(xw, act, erow, ecol, nrm);

  // ---- layer 4: 128 -> 128 ---------------------------------------------
  k_gemm_wmma<128, 128, 128, 1><<<gemmGrid, 256, 0, stream>>>(act, wt4, xw);
  k_seed<<<NT, 128, 0, stream>>>(xw, dinv, b4, act);
  k_scatter<128><<<scat128, 256, 0, stream>>>(xw, act, erow, ecol, nrm);

  // ---- layer 5: 128 -> 128 ---------------------------------------------
  k_gemm_wmma<128, 128, 128, 1><<<gemmGrid, 256, 0, stream>>>(act, wt5, xw);
  k_seed<<<NT, 128, 0, stream>>>(xw, dinv, b5, act);
  k_scatter<128><<<scat128, 256, 0, stream>>>(xw, act, erow, ecol, nrm);

  // ---- features (final ReLU fused) --------------------------------------
  k_feat_cvt<<<NT, 128, 0, stream>>>(act, featb);

  // ---- dueling head ------------------------------------------------------
  constexpr int HEAD_BLOCKS = 64;
  constexpr int KPB = FEATK / HEAD_BLOCKS;   // 8192 (per wave: 1024)
  k_seed_small<<<ceil_div(32 * 32, 256), 256, 0, stream>>>(advp, advb, 32, 32);
  k_head_gemm<32, KPB><<<HEAD_BLOCKS, 256, 0, stream>>>(featb, wtadv, advp);
  k_relu_inplace<<<ceil_div(32 * 32, 256), 256, 0, stream>>>(advp, 32 * 32);

  k_seed_small<<<ceil_div(32 * 64, 256), 256, 0, stream>>>(v1a, v1b, 32, 64);
  k_head_gemm<64, KPB><<<HEAD_BLOCKS, 256, 0, stream>>>(featb, wtv1, v1a);
  k_relu_inplace<<<ceil_div(32 * 64, 256), 256, 0, stream>>>(v1a, 32 * 64);

  k_v2_dense<<<ceil_div(32 * 64, 256), 256, 0, stream>>>(v1a, v2W, v2b, v2a);
  k_v3_dense<<<1, 32, 0, stream>>>(v2a, v3W, v3b, valb);
  k_combine<<<ceil_div(1024, 256), 256, 0, stream>>>(valb, advp, outp);
}
